// LawinAttn_5669356836308
// MI455X (gfx1250) — matrix operand order repaired and checked by reading, not compiled, not run on hardware
//
#include <hip/hip_runtime.h>

// ---------------------------------------------------------------------------
// LawinAttn for MI455X (gfx1250): one workgroup (8 wave32) per batch window.
// Weights are packed f32->bf16 ONCE per launch (prep kernel -> __device__
// buffer, L2-resident), so the hot loops issue only global_load_b128 /
// ds_load_b128 fragment loads + v_wmma_f32_16x16x32_bf16.
// ---------------------------------------------------------------------------

#define LDB 520        // ctx/query token-major ld (elems): 1040B row = 16*65
#define LDP 264        // theta/phi/y token-major ld: 528B row = 16*33
#define LDG 72         // g / attnT ld: 144B row = 16*9
#define LDS_SC 65      // f32 score ld (bank-conflict-free column access)

#define OFF_CTX  0
#define OFF_Q    66560      // 64*520*2
#define OFF_TH   133120
#define OFF_PHI  166912
#define OFF_G    200704
#define OFF_INV  237568
#define OFF_SHF  239616
#define SMEM_BYTES 241664

// packed bf16 weight buffer element offsets
#define WPK_PM   0          // [4][64][64]
#define WPK_G    16384      // [256][512]
#define WPK_TH   147456
#define WPK_PHI  278528
#define WPK_OUT  409600     // [512][256]
#define WPK_N    540672     // total bf16 elements
__device__ unsigned g_wpack[WPK_N / 2];   // 1.03 MB, rewritten every launch

typedef __attribute__((ext_vector_type(16))) __bf16 v16bf;
typedef __attribute__((ext_vector_type(8)))  float  v8f;

union FragB16 { v16bf v; uint4 q[2]; unsigned u32[8]; unsigned short u16[16]; };

// ---- f32 -> bf16 (RNE) via packed hardware converter ----------------------
static __device__ __forceinline__ unsigned pk2(float x, float y) {
#if __has_builtin(__builtin_amdgcn_cvt_pk_bf16_f32)
  auto h = __builtin_amdgcn_cvt_pk_bf16_f32(x, y);   // v_cvt_pk_bf16_f32
  unsigned u; __builtin_memcpy(&u, &h, 4);
  return u;
#else
  union { __bf16 h[2]; unsigned u; } u;
  u.h[0] = (__bf16)x; u.h[1] = (__bf16)y;
  return u.u;
#endif
}
static __device__ __forceinline__ unsigned short f2bf(float f) {
  union { __bf16 h; unsigned short s; } u;
  u.h = (__bf16)f;
  return u.s;
}
static __device__ __forceinline__ float bf2f(unsigned short s) {
  union { unsigned u; float f; } x; x.u = ((unsigned)s) << 16;
  return x.f;
}

static __device__ __forceinline__ v8f wmma_bf16(v16bf a, v16bf b, v8f c) {
  return __builtin_amdgcn_wmma_f32_16x16x32_bf16(false, a, false, b,
                                                 (short)0, c, false, false);
}

// ---- A fragment (16x32 MxK) from row-major bf16 (LDS or global) -----------
// lane<16: M=lane, K = {kb..kb+7, kb+16..kb+23}, kb=0; lane>=16: kb=8.
static __device__ __forceinline__ v16bf fragA(const unsigned short* base, int ld,
                                              int r0, int c0, int lane) {
  FragB16 f;
  const unsigned short* p =
      base + (r0 + (lane & 15)) * ld + c0 + ((lane >> 4) << 3);
  f.q[0] = *(const uint4*)(p);        // K kb..kb+7
  f.q[1] = *(const uint4*)(p + 16);   // K kb+16..kb+23
  return f.v;
}

// A fragment gathered transposed: element (m,k) = base[(k0+k)*ld + m0+m].
static __device__ __forceinline__ v16bf fragAT(const unsigned short* base, int ld,
                                               int k0, int m0, int lane) {
  FragB16 f;
  const int m  = m0 + (lane & 15);
  const int kb = k0 + ((lane >> 4) << 3);
#pragma unroll
  for (int v = 0; v < 8; ++v) {
    const int k = kb + ((v >> 2) << 4) + ((v & 3) << 1);
    f.u16[2 * v]     = base[(k)     * ld + m];
    f.u16[2 * v + 1] = base[(k + 1) * ld + m];
  }
  return f.v;
}

// ---- B fragment (32x16 KxN), element (k,n) = base[n*ld + k] ---------------
// lane<16: N=lane, K = {kb..kb+7, kb+16..kb+23}, kb=k0; lane>=16: kb=k0+8.
static __device__ __forceinline__ v16bf fragBt(const unsigned short* base, int ld,
                                               int k0, int n0, int lane) {
  FragB16 f;
  const unsigned short* p =
      base + (n0 + (lane & 15)) * ld + k0 + ((lane >> 4) << 3);
  f.q[0] = *(const uint4*)(p);
  f.q[1] = *(const uint4*)(p + 16);
  return f.v;
}

// ---- C fragment helpers (v8f: VGPR r -> M=r0+r+8*(lane>=16), N=c0+(lane&15))
static __device__ __forceinline__ v8f ldsCT(const unsigned short* base, int ld,
                                            int r0, int c0, int lane) {
  const unsigned short* p =
      base + (c0 + (lane & 15)) * ld + r0 + ((lane >> 4) << 3);
  union { uint4 q; unsigned short s[8]; } u;
  u.q = *(const uint4*)(p);
  v8f c;
#pragma unroll
  for (int r = 0; r < 8; ++r) c[r] = bf2f(u.s[r]);
  return c;
}
static __device__ __forceinline__ void stCT(unsigned short* base, int ld,
                                            int r0, int c0, int lane, v8f c) {
  unsigned short* p =
      base + (c0 + (lane & 15)) * ld + r0 + ((lane >> 4) << 3);
  union { uint4 q; unsigned u32[4]; } u;
#pragma unroll
  for (int i = 0; i < 4; ++i) u.u32[i] = pk2(c[2 * i], c[2 * i + 1]);
  *(uint4*)(p) = u.q;
}
static __device__ __forceinline__ void stC(unsigned short* base, int ld,
                                           int r0, int c0, int lane, v8f c) {
  const int n  = c0 + (lane & 15);
  const int mb = r0 + ((lane >> 4) << 3);
#pragma unroll
  for (int r = 0; r < 8; ++r) base[(mb + r) * ld + n] = f2bf(c[r]);
}
static __device__ __forceinline__ void stCf32(float* base, int ld,
                                              int r0, int c0, int lane, v8f c) {
  const int n  = c0 + (lane & 15);
  const int mb = r0 + ((lane >> 4) << 3);
#pragma unroll
  for (int r = 0; r < 8; ++r) base[(mb + r) * ld + n] = c[r];
}

// ---------------------------------------------------------------------------
// Prep kernel: pack all weight matrices to bf16 (one v_cvt_pk per pair).
__global__ __launch_bounds__(256)
void pack_weights_kernel(const float* __restrict__ pm_w,
                         const float* __restrict__ g_w,
                         const float* __restrict__ theta_w,
                         const float* __restrict__ phi_w,
                         const float* __restrict__ out_w) {
  const int i = blockIdx.x * 256 + threadIdx.x;   // pair index
  if (i >= WPK_N / 2) return;
  const int e = i << 1;                           // element index
  const float* src;
  int off;
  if (e < WPK_G)        { src = pm_w;    off = e; }
  else if (e < WPK_TH)  { src = g_w;     off = e - WPK_G; }
  else if (e < WPK_PHI) { src = theta_w; off = e - WPK_TH; }
  else if (e < WPK_OUT) { src = phi_w;   off = e - WPK_PHI; }
  else                  { src = out_w;   off = e - WPK_OUT; }
  const float2 q = *(const float2*)(src + off);
  g_wpack[i] = pk2(q.x, q.y);
}

// ---------------------------------------------------------------------------
__global__ __launch_bounds__(256)
void lawin_attn_kernel(const float* __restrict__ query,
                       const float* __restrict__ context,
                       const float* __restrict__ pm_b,
                       const float* __restrict__ g_b,
                       const float* __restrict__ theta_b,
                       const float* __restrict__ phi_b,
                       const float* __restrict__ bn_gamma,
                       const float* __restrict__ bn_beta,
                       const float* __restrict__ bn_mean,
                       const float* __restrict__ bn_var,
                       float* __restrict__ out) {
  extern __shared__ char smem[];
  unsigned short* sCtx = (unsigned short*)(smem + OFF_CTX); // [64][520] tok-major
  unsigned short* sQ   = (unsigned short*)(smem + OFF_Q);   // [64][520] tok-major
  unsigned short* sTh  = (unsigned short*)(smem + OFF_TH);  // [64][264] tok-major
  unsigned short* sPhi = (unsigned short*)(smem + OFF_PHI); // [64][264] tok-major
  unsigned short* sG   = (unsigned short*)(smem + OFF_G);   // [256][72] chan-major
  float*          sInv = (float*)(smem + OFF_INV);          // [512]
  float*          sShf = (float*)(smem + OFF_SHF);          // [512]
  // phase-aliased regions:
  float*          sAttn  = (float*)(smem + OFF_CTX);        // [256][65] f32 scores
  unsigned short* sAttnT = (unsigned short*)(smem + OFF_Q); // [256][72] bf16 softmax
  unsigned short* sY     = (unsigned short*)(smem + OFF_CTX); // [64][264] tok-major

  const unsigned short* wpk = (const unsigned short*)g_wpack;
  const unsigned short* wPM  = wpk + WPK_PM;    // [4][64][64]
  const unsigned short* wG   = wpk + WPK_G;     // [256][512]
  const unsigned short* wTH  = wpk + WPK_TH;
  const unsigned short* wPHI = wpk + WPK_PHI;
  const unsigned short* wOUT = wpk + WPK_OUT;   // [512][256]

  const int tid  = threadIdx.x;
  const int lane = tid & 31;
  const int wid  = tid >> 5;                 // 8 wave32 per workgroup
  const int b    = blockIdx.x;

  const float* ctxG = context + (size_t)b * (512 * 64);
  const float* qG   = query   + (size_t)b * (512 * 64);
  float*       outG = out     + (size_t)b * (512 * 64);

  // Keep the packed weights hot in L2 (global_prefetch_b8).
  for (int i = tid; i < 2048; i += 256)
    __builtin_prefetch(wpk + i * 512, 0, 1);

  // ---- Phase A: stage ctx & query token-major bf16; BN affine precompute ---
  for (int i = tid; i < 256 * 16; i += 256) {
    const int c2 = i >> 4;              // channel pair index (0..255)
    const int t4 = (i & 15) << 2;       // token base (0..60)
    const int c  = c2 << 1;
    const float4 a0 = *(const float4*)(ctxG + (c)     * 64 + t4);
    const float4 a1 = *(const float4*)(ctxG + (c + 1) * 64 + t4);
    *(unsigned*)(sCtx + (t4 + 0) * LDB + c) = pk2(a0.x, a1.x);
    *(unsigned*)(sCtx + (t4 + 1) * LDB + c) = pk2(a0.y, a1.y);
    *(unsigned*)(sCtx + (t4 + 2) * LDB + c) = pk2(a0.z, a1.z);
    *(unsigned*)(sCtx + (t4 + 3) * LDB + c) = pk2(a0.w, a1.w);
    const float4 b0 = *(const float4*)(qG + (c)     * 64 + t4);
    const float4 b1 = *(const float4*)(qG + (c + 1) * 64 + t4);
    *(unsigned*)(sQ + (t4 + 0) * LDB + c) = pk2(b0.x, b1.x);
    *(unsigned*)(sQ + (t4 + 1) * LDB + c) = pk2(b0.y, b1.y);
    *(unsigned*)(sQ + (t4 + 2) * LDB + c) = pk2(b0.z, b1.z);
    *(unsigned*)(sQ + (t4 + 3) * LDB + c) = pk2(b0.w, b1.w);
  }
  for (int o = tid; o < 512; o += 256) {
    const float inv = bn_gamma[o] * rsqrtf(bn_var[o] + 1e-5f);
    sInv[o] = inv;
    sShf[o] = bn_beta[o] - bn_mean[o] * inv;
  }
  __syncthreads();

  // ---- Phase B: position mixing in place: ctx += ctx @ pm_w[h]^T + pm_b ----
  // Wave w owns channels [64w, 64w+63] (inside head w/2): pm_w B tiles are
  // loaded once per token-tile and shared by the 4 A row-blocks.
  {
    const int h     = wid >> 1;
    const int Rbase = wid << 6;
    const unsigned short* pmWh = wPM + h * (64 * 64);
    const float* pmBh = pm_b + h * 64;
    v16bf a[4][2];
#pragma unroll
    for (int blk = 0; blk < 4; ++blk) {
      a[blk][0] = fragAT(sCtx, LDB, 0,  Rbase + (blk << 4), lane);
      a[blk][1] = fragAT(sCtx, LDB, 32, Rbase + (blk << 4), lane);
    }
#pragma unroll
    for (int nt = 0; nt < 4; ++nt) {
      const int c0 = nt << 4;
      const v16bf b0 = fragBt(pmWh, 64, 0,  c0, lane);   // global_load_b128 x2
      const v16bf b1 = fragBt(pmWh, 64, 32, c0, lane);
      const float bias = pmBh[c0 + (lane & 15)];         // bias per token m
#pragma unroll
      for (int blk = 0; blk < 4; ++blk) {
        const int R = Rbase + (blk << 4);
        v8f acc = ldsCT(sCtx, LDB, R, c0, lane);         // residual as C-init
#pragma unroll
        for (int r = 0; r < 8; ++r) acc[r] += bias;
        acc = wmma_bf16(a[blk][0], b0, acc);
        acc = wmma_bf16(a[blk][1], b1, acc);
        stCT(sCtx, LDB, R, c0, lane, acc);
      }
    }
  }
  __syncthreads();

  // ---- Phase C: g/theta/phi projections, N-blocked (A reused by 4 WMMAs) ---
  for (int gr = wid; gr < 48; gr += 8) {
    const int job = gr >> 4;             // 0=g, 1=theta, 2=phi
    const int mt  = gr & 15;
    const unsigned short* W;
    const float* bias;
    const unsigned short* src;
    if (job == 0)      { W = wG;   bias = g_b;     src = sCtx; }
    else if (job == 1) { W = wTH;  bias = theta_b; src = sQ;   }
    else               { W = wPHI; bias = phi_b;   src = sCtx; }
    const int r0 = mt << 4;
    v8f acc[4] = {};
#pragma unroll
    for (int kc = 0; kc < 16; ++kc) {
      const v16bf a = fragA(W, 512, r0, kc << 5, lane);  // global_load_b128 x2
#pragma unroll
      for (int nt = 0; nt < 4; ++nt)
        acc[nt] = wmma_bf16(a, fragBt(src, LDB, kc << 5, nt << 4, lane), acc[nt]);
    }
    const int mb = r0 + ((lane >> 4) << 3);
#pragma unroll
    for (int nt = 0; nt < 4; ++nt) {
#pragma unroll
      for (int r = 0; r < 8; ++r) acc[nt][r] += bias[mb + r];
      if (job == 0)      stC (sG,   LDG, r0, nt << 4, lane, acc[nt]);
      else if (job == 1) stCT(sTh,  LDP, r0, nt << 4, lane, acc[nt]);
      else               stCT(sPhi, LDP, r0, nt << 4, lane, acc[nt]);
    }
  }
  __syncthreads();

  // ---- Phase D: scores S = theta^T @ phi / 8, f32 into LDS -----------------
  for (int t = wid; t < 64; t += 8) {
    const int h = t >> 4, mt = (t & 15) >> 2, nt = t & 3;
    const int hb = h << 6;
    v8f acc = {};
#pragma unroll
    for (int kc = 0; kc < 2; ++kc) {
      const v16bf a  = fragA (sTh,  LDP, mt << 4, hb + (kc << 5), lane);
      const v16bf bm = fragBt(sPhi, LDP, hb + (kc << 5), nt << 4, lane);
      acc = wmma_bf16(a, bm, acc);
    }
#pragma unroll
    for (int r = 0; r < 8; ++r) acc[r] *= 0.125f;       // 1/sqrt(64)
    stCf32(sAttn, LDS_SC, hb + (mt << 4), nt << 4, lane, acc);
  }
  __syncthreads();

  // ---- Phase E: softmax per (h,nq) row; bf16 out is already B-ready --------
  {
    const float* row = sAttn + tid * LDS_SC;            // row = h*64+nq = tid
    float e[64];
    float mx = -3.0e38f;
#pragma unroll
    for (int m = 0; m < 64; ++m) { e[m] = row[m]; mx = fmaxf(mx, e[m]); }
    float s = 0.f;
#pragma unroll
    for (int m = 0; m < 64; ++m) { e[m] = __expf(e[m] - mx); s += e[m]; }
    const float rs = 1.0f / s;
    unsigned short* dst = sAttnT + tid * LDG;           // contiguous in m
#pragma unroll
    for (int m = 0; m < 64; m += 4) {
      uint2 w;
      w.x = pk2(e[m] * rs,     e[m + 1] * rs);
      w.y = pk2(e[m + 2] * rs, e[m + 3] * rs);
      *(uint2*)(dst + m) = w;                           // ds_store_b64
    }
  }
  __syncthreads();

  // ---- Phase F: y = g @ attn^T per head ------------------------------------
  for (int t = wid; t < 64; t += 8) {
    const int h = t >> 4, mt = (t & 15) >> 2, nt = t & 3;
    const int hb = h << 6;
    v8f acc = {};
#pragma unroll
    for (int kc = 0; kc < 2; ++kc) {
      const v16bf a  = fragA (sG, LDG, hb + (mt << 4), kc << 5, lane);
      const v16bf bm = fragBt(sAttnT, LDG, kc << 5, hb + (nt << 4), lane);
      acc = wmma_bf16(a, bm, acc);
    }
    stCT(sY, LDP, hb + (mt << 4), nt << 4, lane, acc);  // tok-major for phase G
  }
  __syncthreads();

  // ---- Phase G: out = BN(out_w @ y) + query, N-blocked ---------------------
  for (int mt = wid; mt < 32; mt += 8) {
    const int r0 = mt << 4;
    v8f acc[4] = {};
#pragma unroll
    for (int kc = 0; kc < 8; ++kc) {
      const v16bf a = fragA(wOUT, 256, r0, kc << 5, lane);
#pragma unroll
      for (int nt = 0; nt < 4; ++nt)
        acc[nt] = wmma_bf16(a, fragBt(sY, LDP, kc << 5, nt << 4, lane), acc[nt]);
    }
    const int mb = r0 + ((lane >> 4) << 3);
#pragma unroll
    for (int nt = 0; nt < 4; ++nt) {
      const int n = (nt << 4) + (lane & 15);
#pragma unroll
      for (int r = 0; r < 8; ++r) {
        const int o = mb + r;
        outG[o * 64 + n] = acc[nt][r] * sInv[o] + sShf[o] + qG[o * 64 + n];
      }
    }
  }
}

// ---------------------------------------------------------------------------
extern "C" void kernel_launch(void* const* d_in, const int* in_sizes, int n_in,
                              void* d_out, int out_size, void* d_ws, size_t ws_size,
                              hipStream_t stream) {
  const float* query   = (const float*)d_in[0];
  const float* context = (const float*)d_in[1];
  const float* pm_w    = (const float*)d_in[2];
  const float* pm_b    = (const float*)d_in[3];
  const float* g_w     = (const float*)d_in[4];
  const float* g_b     = (const float*)d_in[5];
  const float* theta_w = (const float*)d_in[6];
  const float* theta_b = (const float*)d_in[7];
  const float* phi_w   = (const float*)d_in[8];
  const float* phi_b   = (const float*)d_in[9];
  const float* out_w   = (const float*)d_in[10];
  const float* bn_g    = (const float*)d_in[11];
  const float* bn_b    = (const float*)d_in[12];
  const float* bn_m    = (const float*)d_in[13];
  const float* bn_v    = (const float*)d_in[14];
  float* outp = (float*)d_out;

  const int batch = in_sizes[0] / (512 * 64);   // 2048

  // 1) Pack weights to bf16 (deterministic: rewritten from inputs each call).
  const int prep_blocks = (WPK_N / 2 + 255) / 256;
  pack_weights_kernel<<<prep_blocks, 256, 0, stream>>>(
      pm_w, g_w, theta_w, phi_w, out_w);

  // 2) Main fused kernel.
  (void)hipFuncSetAttribute((const void*)lawin_attn_kernel,
                            hipFuncAttributeMaxDynamicSharedMemorySize,
                            SMEM_BYTES);
  lawin_attn_kernel<<<batch, 256, SMEM_BYTES, stream>>>(
      query, context, pm_b, g_b, theta_b, phi_b,
      bn_g, bn_b, bn_m, bn_v, outp);
}
